// Attention_16277926052564
// MI455X (gfx1250) — matrix operand order repaired and checked by reading, not compiled, not run on hardware
//
#include <hip/hip_runtime.h>
#include <stdint.h>

#define B_  64
#define S_  2048
#define DE_ 512
#define DD_ 512

typedef __attribute__((ext_vector_type(16))) _Float16 v16h;
typedef __attribute__((ext_vector_type(8)))  _Float16 v8h;
typedef __attribute__((ext_vector_type(4)))  _Float16 v4h;
typedef __attribute__((ext_vector_type(8)))  float    v8f;

union V16U { v16h v; v8h h[2]; };

__device__ __forceinline__ float fast_tanh(float x) {
    x = fminf(fmaxf(x, -12.0f), 12.0f);
    float e = __expf(2.0f * x);                 // v_exp_f32 (trans, co-executes)
    return (e - 1.0f) * __builtin_amdgcn_rcpf(e + 1.0f);
}

// ---------------------------------------------------------------- W1 -> f16
__global__ __launch_bounds__(256) void k_convert_w1(const float* __restrict__ W1,
                                                    _Float16* __restrict__ W1h) {
    int i = blockIdx.x * 256 + threadIdx.x;     // 65536 float4 total, grid 256
    float4 f = ((const float4*)W1)[i];
    v4h h = { (_Float16)f.x, (_Float16)f.y, (_Float16)f.z, (_Float16)f.w };
    ((v4h*)W1h)[i] = h;
}

// ------------------------------------------------- dec_proj[b,e] = h[b,:]·W2[e,:]
__global__ __launch_bounds__(256) void k_decproj(const float* __restrict__ h,
                                                 const float* __restrict__ W2,
                                                 float* __restrict__ decp) {
    int b = blockIdx.x, t = threadIdx.x;
    __shared__ float hb[DD_];
    hb[t]       = h[b * DD_ + t];
    hb[t + 256] = h[b * DD_ + t + 256];
    __syncthreads();
    for (int e = t; e < DD_; e += 256) {
        const float* w2r = W2 + (size_t)e * DD_;
        float acc = 0.f;
        #pragma unroll 8
        for (int k = 0; k < DD_; ++k) acc = fmaf(hb[k], w2r[k], acc);
        decp[b * DD_ + e] = acc;
    }
}

// ---------------------------------- fused: enc@W1^T -> +dec_proj -> tanh -> ·v
#define LDS_STRIDE 520          // 512 + 8 halves pad (bank-conflict avoidance)
#define LDS_BYTES  (128 * LDS_STRIDE * 2)

__global__ __launch_bounds__(256) void k_scores(const float* __restrict__ enc,
                                                const _Float16* __restrict__ W1h,
                                                const float* __restrict__ decp,
                                                const float* __restrict__ v,
                                                float* __restrict__ scores_out) {
    extern __shared__ _Float16 ldsA[];
    const int tid = threadIdx.x;
    const int bid = blockIdx.x;
    const int b   = bid >> 4;           // 16 s-tiles of 128 rows per batch
    const int s0  = (bid & 15) * 128;

    // Stage enc tile [128 x 512] fp32 -> f16 LDS (read enc exactly once)
    const float* src = enc + ((size_t)(b * S_ + s0)) * DE_;
    #pragma unroll 4
    for (int i = tid; i < 128 * 128; i += 256) {     // float4 granules
        int row = i >> 7;
        int c4  = i & 127;
        float4 f = ((const float4*)src)[(size_t)row * 128 + c4];
        v4h hh = { (_Float16)f.x, (_Float16)f.y, (_Float16)f.z, (_Float16)f.w };
        *(v4h*)(&ldsA[row * LDS_STRIDE + (c4 << 2)]) = hh;
    }
    __syncthreads();

    const int lane = tid & 31;
    const int wv   = tid >> 5;          // 8 waves, each owns 16 rows
    const int lo   = lane & 15;
    const int hi   = lane >> 4;
    const int m0   = wv * 16;

    // A fragment base (ISA 16-bit A 16x32 layout: chunks at hi*8 and hi*8+16)
    const _Float16* Arow = ldsA + (m0 + lo) * LDS_STRIDE + hi * 8;

    float sc[8] = {0.f, 0.f, 0.f, 0.f, 0.f, 0.f, 0.f, 0.f};
    const v8f vzero = {0.f, 0.f, 0.f, 0.f, 0.f, 0.f, 0.f, 0.f};

    #pragma unroll 1
    for (int n0 = 0; n0 < DD_; n0 += 64) {          // 8 d-chunks of 4 N-tiles
        v8f cacc[4];
        cacc[0] = vzero; cacc[1] = vzero; cacc[2] = vzero; cacc[3] = vzero;

        // B fragment: row of W1h (N = d = n0+16t+lo), K chunks at hi*8 and hi*8+16
        const _Float16* Bp0 = W1h + (size_t)(n0 +  0 + lo) * DE_ + hi * 8;
        const _Float16* Bp1 = W1h + (size_t)(n0 + 16 + lo) * DE_ + hi * 8;
        const _Float16* Bp2 = W1h + (size_t)(n0 + 32 + lo) * DE_ + hi * 8;
        const _Float16* Bp3 = W1h + (size_t)(n0 + 48 + lo) * DE_ + hi * 8;

        #pragma unroll
        for (int k = 0; k < DE_; k += 32) {
            V16U a;
            a.h[0] = *(const v8h*)(Arow + k);
            a.h[1] = *(const v8h*)(Arow + k + 16);
            V16U u0, u1, u2, u3;
            u0.h[0] = *(const v8h*)(Bp0 + k); u0.h[1] = *(const v8h*)(Bp0 + k + 16);
            u1.h[0] = *(const v8h*)(Bp1 + k); u1.h[1] = *(const v8h*)(Bp1 + k + 16);
            u2.h[0] = *(const v8h*)(Bp2 + k); u2.h[1] = *(const v8h*)(Bp2 + k + 16);
            u3.h[0] = *(const v8h*)(Bp3 + k); u3.h[1] = *(const v8h*)(Bp3 + k + 16);
            v16h b0 = u0.v, b1 = u1.v, b2 = u2.v, b3 = u3.v;
            cacc[0] = __builtin_amdgcn_wmma_f32_16x16x32_f16(false, a.v, false, b0,
                                                             (short)0, cacc[0], false, false);
            cacc[1] = __builtin_amdgcn_wmma_f32_16x16x32_f16(false, a.v, false, b1,
                                                             (short)0, cacc[1], false, false);
            cacc[2] = __builtin_amdgcn_wmma_f32_16x16x32_f16(false, a.v, false, b2,
                                                             (short)0, cacc[2], false, false);
            cacc[3] = __builtin_amdgcn_wmma_f32_16x16x32_f16(false, a.v, false, b3,
                                                             (short)0, cacc[3], false, false);
        }

        // Epilogue: C element (r,lane) = enc_proj[row=r+8*hi, col=n0+16t+lo]
        #pragma unroll
        for (int t = 0; t < 4; ++t) {
            int d    = n0 + 16 * t + lo;
            float vd = v[d];
            float dp = decp[b * DD_ + d];
            #pragma unroll
            for (int r = 0; r < 8; ++r)
                sc[r] = fmaf(vd, fast_tanh(cacc[t][r] + dp), sc[r]);
        }
    }

    // Reduce across the 16 lanes of each half (xor 1..8 never crosses bit 4)
    #pragma unroll
    for (int m = 1; m <= 8; m <<= 1) {
        #pragma unroll
        for (int r = 0; r < 8; ++r) sc[r] += __shfl_xor(sc[r], m, 32);
    }
    if (lo == 0) {
        int base = b * S_ + s0 + m0 + hi * 8;   // rows r+8*hi
        #pragma unroll
        for (int r = 0; r < 8; ++r) scores_out[base + r] = sc[r];
    }
}

// ------------------------------------------------------------ softmax (in place)
__global__ __launch_bounds__(256) void k_softmax(float* __restrict__ attn,
                                                 const uint8_t* __restrict__ mask) {
    int b = blockIdx.x, t = threadIdx.x;
    float* row = attn + (size_t)b * S_;
    const uint8_t* mk = mask + (size_t)b * S_;

    float vals[8];
    float mx = -__builtin_inff();
    #pragma unroll
    for (int i = 0; i < 8; ++i) {
        int s = t + i * 256;
        float x = row[s];
        if (mk[s]) x = -__builtin_inff();
        vals[i] = x;
        mx = fmaxf(mx, x);
    }
    #pragma unroll
    for (int m = 1; m < 32; m <<= 1) mx = fmaxf(mx, __shfl_xor(mx, m, 32));
    __shared__ float redm[8];
    if ((t & 31) == 0) redm[t >> 5] = mx;
    __syncthreads();
    mx = redm[0];
    #pragma unroll
    for (int w = 1; w < 8; ++w) mx = fmaxf(mx, redm[w]);

    float sum = 0.f;
    #pragma unroll
    for (int i = 0; i < 8; ++i) { float e = __expf(vals[i] - mx); vals[i] = e; sum += e; }
    #pragma unroll
    for (int m = 1; m < 32; m <<= 1) sum += __shfl_xor(sum, m, 32);
    __shared__ float reds[8];
    if ((t & 31) == 0) reds[t >> 5] = sum;
    __syncthreads();
    sum = reds[0];
    #pragma unroll
    for (int w = 1; w < 8; ++w) sum += reds[w];

    float inv = __builtin_amdgcn_rcpf(sum);
    #pragma unroll
    for (int i = 0; i < 8; ++i) row[t + i * 256] = vals[i] * inv;
}

// ---------------------------------------- context = w @ enc  (two-pass, no atomics)
__global__ __launch_bounds__(256) void k_ctx_partial(const float* __restrict__ enc,
                                                     const float* __restrict__ attn,
                                                     float* __restrict__ part) {
    int bc = blockIdx.x;                 // 512 = 64 batches * 8 s-chunks
    int b = bc >> 3, c = bc & 7;
    int t = threadIdx.x;
    const float* E = enc + ((size_t)(b * S_ + c * 256)) * DE_;
    const float* W = attn + (size_t)b * S_ + c * 256;
    float a0 = 0.f, a1 = 0.f;
    #pragma unroll 4
    for (int s = 0; s < 256; ++s) {
        float w = W[s];
        a0 = fmaf(w, E[(size_t)s * DE_ + t],       a0);
        a1 = fmaf(w, E[(size_t)s * DE_ + t + 256], a1);
    }
    part[((size_t)c * B_ + b) * DE_ + t]       = a0;
    part[((size_t)c * B_ + b) * DE_ + t + 256] = a1;
}

__global__ __launch_bounds__(256) void k_ctx_reduce(const float* __restrict__ part,
                                                    float* __restrict__ ctx) {
    int b = blockIdx.x, t = threadIdx.x;
    for (int e = t; e < DE_; e += 256) {
        float a = 0.f;
        #pragma unroll
        for (int c = 0; c < 8; ++c) a += part[((size_t)c * B_ + b) * DE_ + e];
        ctx[(size_t)b * DE_ + e] = a;
    }
}

// ------------------------------------------------------------------ launcher
extern "C" void kernel_launch(void* const* d_in, const int* in_sizes, int n_in,
                              void* d_out, int out_size, void* d_ws, size_t ws_size,
                              hipStream_t stream) {
    const float*   dec_h = (const float*)d_in[0];     // [B,DD]
    const float*   enc   = (const float*)d_in[1];     // [B,S,DE]
    const uint8_t* mask  = (const uint8_t*)d_in[2];   // [B,S] bool
    const float*   W1    = (const float*)d_in[3];     // [DD,DE]
    const float*   W2    = (const float*)d_in[4];     // [DD,DD]
    const float*   v     = (const float*)d_in[5];     // [DD]

    float* attn = (float*)d_out;                      // [B,S]
    float* ctx  = (float*)d_out + (size_t)B_ * S_;    // [B,DE]

    char* ws = (char*)d_ws;                           // needs ~1.7 MB
    _Float16* W1h  = (_Float16*)ws;                               // 512 KB
    float*    decp = (float*)(ws + 524288);                       // 128 KB
    float*    part = (float*)(ws + 524288 + 131072);              // 1 MB

    k_convert_w1 <<<256,  256, 0,         stream>>>(W1, W1h);
    k_decproj    <<<B_,   256, 0,         stream>>>(dec_h, W2, decp);
    k_scores     <<<1024, 256, LDS_BYTES, stream>>>(enc, W1h, decp, v, attn);
    k_softmax    <<<B_,   256, 0,         stream>>>(attn, mask);
    k_ctx_partial<<<512,  256, 0,         stream>>>(enc, attn, part);
    k_ctx_reduce <<<B_,   256, 0,         stream>>>(part, ctx);
}
